// GNN_53446573031871
// MI455X (gfx1250) — matrix operand order repaired and checked by reading, not compiled, not run on hardware
//
#include <hip/hip_runtime.h>

// ---------------------------------------------------------------------------
// 2-layer GCN (copy_u + segment_sum aggregation, then dense linear + bias)
// B=4, N=20000, E=320000, F: 128 -> 256 -> 128
//
// Pipeline (all on `stream`):
//   prep : W1 (128x256 f32) -> W1t (256x128 bf16), W2 -> W2t (128x256 bf16)
//   zero bufA ; scatter_add<128>(features) -> bufA
//   gemm<K=128,FOUT=256,RELU>(bufA, W1t, b1) -> bufB
//   zero bufA ; scatter_add<256>(bufB)     -> bufA
//   gemm<K=256,FOUT=128>(bufA, W2t, b2)    -> d_out
//
// Workspace layout (needs ~164 MB):
//   [0,      64K)   W1t bf16
//   [64K,   128K)   W2t bf16
//   [128K,  128K+81.92MB)  bufA f32  (80000 x 256 max)
//   [...,   +81.92MB)      bufB f32  (80000 x 256)
// ---------------------------------------------------------------------------

typedef __attribute__((ext_vector_type(16))) __bf16 v16bf;
typedef __attribute__((ext_vector_type(8)))  __bf16 v8bf;
typedef __attribute__((ext_vector_type(8)))  float  v8f;

__device__ __forceinline__ void atomic_add_f32(float* p, float v) {
    // lowers to global_atomic_add_f32 (no CAS loop)
    unsafeAtomicAdd(p, v);
}

// --------------------------- zero fill -------------------------------------
__global__ void GNN_zero_kernel(float* __restrict__ p, size_t n) {
    size_t i = (size_t)blockIdx.x * blockDim.x + threadIdx.x;
    const size_t stride = (size_t)gridDim.x * blockDim.x;
    for (; i < n; i += stride) p[i] = 0.0f;
}

// ------------------- W transpose + f32 -> bf16 convert ---------------------
// W: [K, FOUT] f32 row-major  ->  Wt: [FOUT, K] bf16 row-major
__global__ void GNN_transpose_w_kernel(const float* __restrict__ W,
                                       __bf16* __restrict__ Wt,
                                       int K, int FOUT) {
    const int idx = blockIdx.x * blockDim.x + threadIdx.x;
    if (idx >= K * FOUT) return;
    const int n = idx / K;
    const int k = idx - n * K;
    Wt[idx] = (__bf16)W[k * FOUT + n];
}

// --------------------------- edge scatter-add ------------------------------
// One wave (32 lanes) per edge; lane handles F/32 floats via float4 loads
// and 4 hardware f32 atomics per float4. grid = (E/8, B), block = 256.
template <int F>
__global__ __launch_bounds__(256) void GNN_scatter_add_kernel(
    const float* __restrict__ feat,   // [B, N, F]
    const int*   __restrict__ src,    // [B, E]
    const int*   __restrict__ dst,    // [B, E]
    float*       __restrict__ out,    // [B, N, F] (pre-zeroed)
    int N, int E) {
    const int b    = blockIdx.y;
    const int wave = threadIdx.x >> 5;
    const int lane = threadIdx.x & 31;
    const int e    = blockIdx.x * 8 + wave;   // E divisible by 8

    const int s = src[(size_t)b * E + e];
    const int d = dst[(size_t)b * E + e];

    const float4* frow = (const float4*)(feat + (size_t)b * N * F + (size_t)s * F);
    float*        orow = out + (size_t)b * N * F + (size_t)d * F;

#pragma unroll
    for (int j = 0; j < F / 128; ++j) {
        const int vi = j * 32 + lane;         // float4 index within row
        const float4 v = frow[vi];
        atomic_add_f32(orow + vi * 4 + 0, v.x);
        atomic_add_f32(orow + vi * 4 + 1, v.y);
        atomic_add_f32(orow + vi * 4 + 2, v.z);
        atomic_add_f32(orow + vi * 4 + 3, v.w);
    }
}

// --------------------------- WMMA GEMM + bias (+ReLU) ----------------------
// out[r, :] = A[r, :] @ W + bias   (A: [rows,K] f32, W given as Wt: [FOUT,K] bf16)
// One block = 16 rows; A tile staged to LDS as bf16 with +8 element pad
// (row stride 272B / 528B -> conflict-free ds_load_b128 fragment reads).
// 8 waves sweep FOUT/16 column tiles; K-loop of 32 with
// v_wmma_f32_16x16x32_bf16 accumulating in f32.
template <int K, int FOUT, bool RELU>
__global__ __launch_bounds__(256) void GNN_gemm_bias_kernel(
    const float*  __restrict__ A,
    const __bf16* __restrict__ Wt,
    const float*  __restrict__ bias,
    float*        __restrict__ out) {
    constexpr int LDA = K + 8;                // bf16 elements per LDS row
    __shared__ __bf16 ldsA[16 * LDA];

    const int tid  = threadIdx.x;
    const int wave = tid >> 5;
    const int lane = tid & 31;
    const size_t row0 = (size_t)blockIdx.x * 16;

    // cooperative load of 16 x K f32 tile -> bf16 LDS (coalesced)
#pragma unroll
    for (int idx = tid; idx < 16 * K; idx += 256) {
        const int r = idx / K;                // K is a power of two
        const int k = idx & (K - 1);
        ldsA[r * LDA + k] = (__bf16)A[(row0 + r) * K + k];
    }
    __syncthreads();

    // ISA 7.12.2 16-bit fragment layouts (wave32):
    // A (16x32): lane 0-15 -> M=lane, K-halves {0..7, 16..23};
    //            lane 16-31 -> M=lane-16, K-halves {8..15, 24..31}
    // B (32x16): lane 0-15 -> N=lane, K {0..7, 16..23}; lane 16-31 -> N=lane-16, K {8..15, 24..31}
    const int rowA  = lane & 15;
    const int kOffA = (lane >> 4) << 3;       // 0 or 8
    const int colB  = lane & 15;
    const int kOffB = (lane >> 4) << 3;       // 0 or 8
    const int mBase = (lane >> 4) << 3;       // C/D: M = mBase + vgpr

    const __bf16* aRow = &ldsA[rowA * LDA + kOffA];

    for (int ct = wave; ct < FOUT / 16; ct += 8) {
        const int col0 = ct * 16;
        const __bf16* bRow = &Wt[(size_t)(col0 + colB) * K + kOffB];

        v8f acc = {0.f, 0.f, 0.f, 0.f, 0.f, 0.f, 0.f, 0.f};
#pragma unroll
        for (int kc = 0; kc < K; kc += 32) {
            const v8bf a_lo = *(const v8bf*)(aRow + kc);        // K = +0..7
            const v8bf a_hi = *(const v8bf*)(aRow + kc + 16);   // K = +16..23
            const v16bf aF = __builtin_shufflevector(
                a_lo, a_hi, 0, 1, 2, 3, 4, 5, 6, 7, 8, 9, 10, 11, 12, 13, 14, 15);

            const v8bf b_lo = *(const v8bf*)(bRow + kc);        // K = +0..7
            const v8bf b_hi = *(const v8bf*)(bRow + kc + 16);   // K = +16..23
            const v16bf bF = __builtin_shufflevector(
                b_lo, b_hi, 0, 1, 2, 3, 4, 5, 6, 7, 8, 9, 10, 11, 12, 13, 14, 15);

            acc = __builtin_amdgcn_wmma_f32_16x16x32_bf16(
                /*neg_a=*/false, aF, /*neg_b=*/false, bF,
                /*c_mod=*/(short)0, acc, /*reuse_a=*/false, /*reuse_b=*/false);
        }

        const float bv = bias[col0 + colB];
#pragma unroll
        for (int r = 0; r < 8; ++r) {
            float v = acc[r] + bv;
            if (RELU) v = v > 0.0f ? v : 0.0f;
            out[(row0 + mBase + r) * FOUT + col0 + colB] = v;
        }
    }
}

// ---------------------------------------------------------------------------
extern "C" void kernel_launch(void* const* d_in, const int* in_sizes, int n_in,
                              void* d_out, int out_size, void* d_ws, size_t ws_size,
                              hipStream_t stream) {
    (void)in_sizes; (void)n_in; (void)out_size; (void)ws_size;

    constexpr int B = 4, N = 20000, E = 320000;
    constexpr int ROWS = B * N;               // 80000, divisible by 16

    const float* features = (const float*)d_in[0];   // [B,N,128]
    const int*   src      = (const int*)  d_in[1];   // [B,E]
    const int*   dst      = (const int*)  d_in[2];   // [B,E]
    const float* W1       = (const float*)d_in[3];   // [128,256]
    const float* b1       = (const float*)d_in[4];   // [256]
    const float* W2       = (const float*)d_in[5];   // [256,128]
    const float* b2       = (const float*)d_in[6];   // [128]
    float*       out      = (float*)d_out;           // [B,N,128]

    char* ws = (char*)d_ws;
    __bf16* W1t = (__bf16*)(ws);                         //  64 KB
    __bf16* W2t = (__bf16*)(ws + 65536);                 //  64 KB
    float*  bufA = (float*)(ws + 131072);                // 80000*256 f32
    float*  bufB = (float*)(ws + 131072 + (size_t)ROWS * 256 * 4);

    // weight transpose + bf16 convert
    GNN_transpose_w_kernel<<<(128 * 256 + 255) / 256, 256, 0, stream>>>(W1, W1t, 128, 256);
    GNN_transpose_w_kernel<<<(256 * 128 + 255) / 256, 256, 0, stream>>>(W2, W2t, 256, 128);

    // ---- layer 1: aggregate 128-d, then 128->256 GEMM + bias + ReLU ----
    GNN_zero_kernel<<<2048, 256, 0, stream>>>(bufA, (size_t)ROWS * 128);
    {
        dim3 grid(E / 8, B);
        GNN_scatter_add_kernel<128><<<grid, 256, 0, stream>>>(features, src, dst, bufA, N, E);
    }
    GNN_gemm_bias_kernel<128, 256, true><<<ROWS / 16, 256, 0, stream>>>(bufA, W1t, b1, bufB);

    // ---- layer 2: aggregate 256-d, then 256->128 GEMM + bias ----
    GNN_zero_kernel<<<4096, 256, 0, stream>>>(bufA, (size_t)ROWS * 256);
    {
        dim3 grid(E / 8, B);
        GNN_scatter_add_kernel<256><<<grid, 256, 0, stream>>>(bufB, src, dst, bufA, N, E);
    }
    GNN_gemm_bias_kernel<256, 128, false><<<ROWS / 16, 256, 0, stream>>>(bufA, W2t, b2, out);
}
